// EdgeConv_10359461118325
// MI455X (gfx1250) — matrix operand order repaired and checked by reading, not compiled, not run on hardware
//
#include <hip/hip_runtime.h>
#include <hip/hip_bf16.h>
#include <math.h>

typedef __attribute__((ext_vector_type(16))) _Float16 v16h;
typedef __attribute__((ext_vector_type(8)))  _Float16 v8h;
typedef __attribute__((ext_vector_type(8)))  float    v8f;

// ---------- helpers: order-preserving float<->uint encoding for atomicMax ----
__device__ __forceinline__ unsigned enc_f32(float f) {
    unsigned b = __float_as_uint(f);
    return b ^ ((((int)b) >> 31) | 0x80000000u);
}
__device__ __forceinline__ float dec_f32(unsigned u) {
    unsigned b = (u & 0x80000000u) ? (u ^ 0x80000000u) : ~u;
    return __uint_as_float(b);
}

// ---------- 1) node transform: h = x@W_gat, a_src = h.att_src, a_dst = h.att_dst
__global__ void node_h_kernel(const float* __restrict__ x,
                              const float* __restrict__ W_gat,
                              const float* __restrict__ att_src,
                              const float* __restrict__ att_dst,
                              float* __restrict__ h,
                              float* __restrict__ a_src,
                              float* __restrict__ a_dst, int N) {
    __shared__ float Ws[80], as_s[16], ad_s[16];
    int tid = threadIdx.x;
    if (tid < 80) Ws[tid] = W_gat[tid];
    if (tid < 16) { as_s[tid] = att_src[tid]; ad_s[tid] = att_dst[tid]; }
    __syncthreads();
    int node = blockIdx.x * blockDim.x + tid;
    if (node >= N) return;
    float xi[5];
#pragma unroll
    for (int k = 0; k < 5; ++k) xi[k] = x[node * 5 + k];
    float hs = 0.f, hd = 0.f;
#pragma unroll
    for (int j = 0; j < 16; ++j) {
        float v = 0.f;
#pragma unroll
        for (int k = 0; k < 5; ++k) v += xi[k] * Ws[k * 16 + j];
        h[node * 16 + j] = v;
        hs += v * as_s[j];
        hd += v * ad_s[j];
    }
    a_src[node] = hs;
    a_dst[node] = hd;
}

// ---------- 2) fold W_edge @ att_edge -> weff[2]
__global__ void weff_kernel(const float* __restrict__ W_edge,
                            const float* __restrict__ att_edge,
                            float* __restrict__ weff) {
    if (threadIdx.x == 0) {
        float w0 = 0.f, w1 = 0.f;
#pragma unroll
        for (int j = 0; j < 16; ++j) {
            w0 += W_edge[j]      * att_edge[j];
            w1 += W_edge[16 + j] * att_edge[j];
        }
        weff[0] = w0; weff[1] = w1;
    }
}

// ---------- 3) alpha = leaky_relu(...), atomic segmented max over dst
__global__ void alpha_max_kernel(const int* __restrict__ src, const int* __restrict__ dst,
                                 const float* __restrict__ edge_attr,
                                 const float* __restrict__ a_src, const float* __restrict__ a_dst,
                                 const float* __restrict__ weff,
                                 float* __restrict__ alpha, unsigned* __restrict__ nodemax, int E) {
    int e = blockIdx.x * blockDim.x + threadIdx.x;
    if (e >= E) return;
    int s = src[e], d = dst[e];
    float a = a_src[s] + a_dst[d] + edge_attr[e * 2] * weff[0] + edge_attr[e * 2 + 1] * weff[1];
    a = (a > 0.f) ? a : 0.2f * a;   // leaky_relu slope 0.2
    alpha[e] = a;
    atomicMax(&nodemax[d], enc_f32(a));
}

// ---------- 4) ex = exp(alpha - max[dst]); atomic segment sum
__global__ void exp_sum_kernel(const int* __restrict__ dst,
                               float* __restrict__ alpha,       // in: alpha, out: ex
                               const unsigned* __restrict__ nodemax,
                               float* __restrict__ denom, int E) {
    int e = blockIdx.x * blockDim.x + threadIdx.x;
    if (e >= E) return;
    int d = dst[e];
    float ex = __expf(alpha[e] - dec_f32(nodemax[d]));
    alpha[e] = ex;
    atomicAdd(&denom[d], ex);
}

// ---------- 5) out[dst] += h[src] * (ex / denom[dst])
__global__ void aggregate_kernel(const int* __restrict__ src, const int* __restrict__ dst,
                                 const float* __restrict__ ex, const float* __restrict__ denom,
                                 const float* __restrict__ h, float* __restrict__ acc, int E) {
    int e = blockIdx.x * blockDim.x + threadIdx.x;
    if (e >= E) return;
    int s = src[e], d = dst[e];
    float a = ex[e] / (denom[d] + 1e-16f);
#pragma unroll
    for (int j = 0; j < 16; ++j)
        atomicAdd(&acc[d * 16 + j], h[s * 16 + j] * a);
}

// ---------- 6) xh = relu(acc + b_gat)
__global__ void relu_bias_kernel(const float* __restrict__ acc, const float* __restrict__ b_gat,
                                 float* __restrict__ xh, int NT) {
    int i = blockIdx.x * blockDim.x + threadIdx.x;
    if (i >= NT) return;
    xh[i] = fmaxf(acc[i] + b_gat[i & 15], 0.f);
}

// ---------- 7) edge MLP via WMMA
// feat[16x64(f16)] @ W1[64x64(f16)] (8 WMMAs) -> +b1, relu -> @W2[64x2] via
// shfl butterfly reduction -> +b2, sigmoid.
__global__ void __launch_bounds__(256)
edge_mlp_wmma_kernel(const float* __restrict__ xh,
                     const int* __restrict__ src, const int* __restrict__ dst,
                     const float* __restrict__ edge_attr,
                     const float* __restrict__ W1, const float* __restrict__ b1,
                     const float* __restrict__ W2, const float* __restrict__ b2,
                     float* __restrict__ ef, int E) {
    // B fragments pre-swizzled to lane layout k = 16*half + i (same as A): one contiguous v16h per lane.
    __shared__ _Float16 W1f[2][4][32][16];   // [kt][nt][lane][i]  (8 KB)
    __shared__ _Float16 feat[8][16][64];     // per-wave edge tile (16 KB)

    int tid  = threadIdx.x;
    int lane = tid & 31;
    int wid  = tid >> 5;

    // ---- fill W1f: thread tid builds fragment (kt,nt,lane_) as one v16h store
    {
        int lane_ = tid & 31, nt = (tid >> 5) & 3, kt = tid >> 7;
        int kh_ = lane_ >> 4, n_ = (lane_ & 15) + nt * 16;
        v16h v;
#pragma unroll
        for (int i = 0; i < 16; ++i) {
            int k = 32 * kt + 16 * kh_ + i;                 // K index (padded to 64)
            v[i] = (_Float16)((k < 34) ? W1[k * 64 + n_] : 0.f);
        }
        *(v16h*)&W1f[kt][nt][lane_][0] = v;
    }

    // ---- build feature tile (f16): 4 x ds_store_b128 per lane
    int tile = blockIdx.x * 8 + wid;
    int e0   = tile * 16;
    {
        int m = lane & 15;
        int e = e0 + m; if (e >= E) e = E - 1;
        int s = src[e], d = dst[e];
        if (lane < 16) {                                    // cols 0..31
            v8h c0, c1, c2, c3;
#pragma unroll
            for (int j = 0; j < 8; ++j) c0[j] = (_Float16)xh[s * 16 + j];
#pragma unroll
            for (int j = 0; j < 8; ++j) c1[j] = (_Float16)xh[s * 16 + 8 + j];
            c2[0] = (_Float16)edge_attr[e * 2];
            c2[1] = (_Float16)edge_attr[e * 2 + 1];
#pragma unroll
            for (int j = 0; j < 6; ++j) c2[2 + j] = (_Float16)xh[d * 16 + j];
#pragma unroll
            for (int j = 0; j < 8; ++j) c3[j] = (_Float16)xh[d * 16 + 6 + j];
            *(v8h*)&feat[wid][m][0]  = c0;
            *(v8h*)&feat[wid][m][8]  = c1;
            *(v8h*)&feat[wid][m][16] = c2;
            *(v8h*)&feat[wid][m][24] = c3;
        } else {                                            // cols 32..63 (mostly pad)
            v8h c4 = {};
            c4[0] = (_Float16)xh[d * 16 + 14];
            c4[1] = (_Float16)xh[d * 16 + 15];
            v8h z = {};
            *(v8h*)&feat[wid][m][32] = c4;
            *(v8h*)&feat[wid][m][40] = z;
            *(v8h*)&feat[wid][m][48] = z;
            *(v8h*)&feat[wid][m][56] = z;
        }
    }
    __syncthreads();

    // ---- A fragments: two contiguous 8-half runs per lane, per K-step
    int half = lane >> 4, m = lane & 15, nlo = lane & 15;
    v8h r0 = *(const v8h*)&feat[wid][m][16 * half];
    v8h r1 = *(const v8h*)&feat[wid][m][8 + 16 * half];
    v8h r2 = *(const v8h*)&feat[wid][m][32 + 16 * half];
    v8h r3 = *(const v8h*)&feat[wid][m][40 + 16 * half];
    v16h a0 = __builtin_shufflevector(r0, r1, 0,1,2,3,4,5,6,7,8,9,10,11,12,13,14,15);
    v16h a1 = __builtin_shufflevector(r2, r3, 0,1,2,3,4,5,6,7,8,9,10,11,12,13,14,15);

    // ---- 4 N-tiles x 2 K-steps: 8 WMMAs, B frags as single v16h LDS loads
    v8f acc[4];
#pragma unroll
    for (int nt = 0; nt < 4; ++nt) {
        v16h b0 = *(const v16h*)&W1f[0][nt][lane][0];
        v16h b1v = *(const v16h*)&W1f[1][nt][lane][0];
        v8f c = {};
        c = __builtin_amdgcn_wmma_f32_16x16x32_f16(false, a0, false, b0,  (short)0, c, false, false);
        c = __builtin_amdgcn_wmma_f32_16x16x32_f16(false, a1, false, b1v, (short)0, c, false, false);
        acc[nt] = c;
    }

    // ---- second GEMM 16x64 @ 64x2 without LDS round-trip.
    // C layout: lane (half,nlo) + VGPR r holds D[row = r + 8*half][col = nt*16 + nlo].
    // Each half-group of 16 lanes holds all 64 columns of its 8 rows (4 cols/lane)
    // -> per-row dot = lane partials + 4-step shfl_xor butterfly over 16 lanes.
    float w20[4], w21[4], b1v4[4];
#pragma unroll
    for (int nt = 0; nt < 4; ++nt) {
        int col = nt * 16 + nlo;
        w20[nt] = W2[col * 2 + 0];
        w21[nt] = W2[col * 2 + 1];
        b1v4[nt] = b1[col];
    }
    float bb0 = b2[0], bb1 = b2[1];
#pragma unroll
    for (int r = 0; r < 8; ++r) {
        float p0 = 0.f, p1 = 0.f;
#pragma unroll
        for (int nt = 0; nt < 4; ++nt) {
            float hv = fmaxf(acc[nt][r] + b1v4[nt], 0.f);
            p0 += hv * w20[nt];
            p1 += hv * w21[nt];
        }
#pragma unroll
        for (int off = 1; off < 16; off <<= 1) {
            p0 += __shfl_xor(p0, off, 32);
            p1 += __shfl_xor(p1, off, 32);
        }
        int row = r + 8 * half;
        int e   = e0 + row;
        if (nlo == 0 && e < E) {
            ef[e * 2 + 0] = 1.f / (1.f + __expf(-(p0 + bb0)));
            ef[e * 2 + 1] = 1.f / (1.f + __expf(-(p1 + bb1)));
        }
    }
}

// ---------- 8) per-block valid counts
__global__ void count_kernel(const int* __restrict__ src, const int* __restrict__ dst,
                             const unsigned char* __restrict__ lab,
                             int* __restrict__ bcounts, int E) {
    __shared__ int wsum[8];
    int tid = threadIdx.x, lane = tid & 31, wid = tid >> 5;
    int e = blockIdx.x * 256 + tid;
    int valid = 0;
    if (e < E) { int s = src[e], d = dst[e]; valid = (lab[s] && lab[d]) ? 1 : 0; }
    unsigned long long mask = __ballot(valid);
    if (lane == 0) wsum[wid] = __popcll(mask);
    __syncthreads();
    if (tid == 0) {
        int r = 0;
#pragma unroll
        for (int i = 0; i < 8; ++i) r += wsum[i];
        bcounts[blockIdx.x] = r;
    }
}

// ---------- 9) exclusive scan of block counts
__global__ void scan_kernel(const int* __restrict__ bcounts, int* __restrict__ boffs, int nb) {
    if (threadIdx.x == 0 && blockIdx.x == 0) {
        int run = 0;
        for (int i = 0; i < nb; ++i) { boffs[i] = run; run += bcounts[i]; }
        boffs[nb] = run;
    }
}

// ---------- 10) stable compaction: write selected edges + sigmoids
__global__ void compact_kernel(const int* __restrict__ src, const int* __restrict__ dst,
                               const unsigned char* __restrict__ lab,
                               const float* __restrict__ ef,
                               const int* __restrict__ boffs,
                               float* __restrict__ out, int E, int K) {
    __shared__ int wsum[8], woff[8];
    int tid = threadIdx.x, lane = tid & 31, wid = tid >> 5;
    int e = blockIdx.x * 256 + tid;
    int valid = 0, s = 0, d = 0;
    if (e < E) { s = src[e]; d = dst[e]; valid = (lab[s] && lab[d]) ? 1 : 0; }
    unsigned long long mask = __ballot(valid);
    int intra = __popcll(mask & ((1ull << lane) - 1ull));
    if (lane == 0) wsum[wid] = __popcll(mask);
    __syncthreads();
    if (tid == 0) {
        int r = 0;
#pragma unroll
        for (int i = 0; i < 8; ++i) { woff[i] = r; r += wsum[i]; }
    }
    __syncthreads();
    if (valid) {
        int pos = boffs[blockIdx.x] + woff[wid] + intra;
        if (pos < K) {
            out[pos]             = (float)s;          // edges[0][idx]
            out[K + pos]         = (float)d;          // edges[1][idx]
            out[2 * K + 2 * pos]     = ef[e * 2];     // sigmoid(ef)[idx]
            out[2 * K + 2 * pos + 1] = ef[e * 2 + 1];
        }
    }
}

extern "C" void kernel_launch(void* const* d_in, const int* in_sizes, int n_in,
                              void* d_out, int out_size, void* d_ws, size_t ws_size,
                              hipStream_t stream) {
    const float* x          = (const float*)d_in[0];
    const int*   edges      = (const int*)  d_in[1];
    const float* edge_attr  = (const float*)d_in[2];
    const unsigned char* lab= (const unsigned char*)d_in[3];
    const float* W_gat      = (const float*)d_in[4];
    const float* att_src    = (const float*)d_in[5];
    const float* att_dst    = (const float*)d_in[6];
    const float* W_edge     = (const float*)d_in[7];
    const float* att_edge   = (const float*)d_in[8];
    const float* b_gat      = (const float*)d_in[9];
    const float* W1         = (const float*)d_in[10];
    const float* b1         = (const float*)d_in[11];
    const float* W2         = (const float*)d_in[12];
    const float* b2         = (const float*)d_in[13];
    float* out = (float*)d_out;

    const int N = in_sizes[0] / 5;
    const int E = in_sizes[1] / 2;
    const int* src = edges;
    const int* dst = edges + E;
    const int nb = (E + 255) / 256;
    const int K = out_size / 4;   // out = [2,K] edges + [K,2] sigmoids

    // workspace layout
    char* ws = (char*)d_ws;
    size_t off = 0;
    float*    h       = (float*)(ws + off); off += (size_t)N * 16 * 4;
    float*    xh      = (float*)(ws + off); off += (size_t)N * 16 * 4;
    float*    a_src   = (float*)(ws + off); off += (size_t)N * 4;
    float*    a_dst   = (float*)(ws + off); off += (size_t)N * 4;
    float*    alpha   = (float*)(ws + off); off += (size_t)E * 4;
    float*    denom   = (float*)(ws + off); off += (size_t)N * 4;
    unsigned* nodemax = (unsigned*)(ws + off); off += (size_t)N * 4;
    float*    acc     = (float*)(ws + off); off += (size_t)N * 16 * 4;
    float*    ef      = (float*)(ws + off); off += (size_t)E * 2 * 4;
    float*    weff    = (float*)(ws + off); off += 16;
    int*      bcounts = (int*)(ws + off);   off += (size_t)nb * 4;
    int*      boffs   = (int*)(ws + off);   off += (size_t)(nb + 1) * 4;

    // zero-init denom / nodemax / acc (contiguous) every call
    hipMemsetAsync(denom, 0, (size_t)N * 4 * 2 + (size_t)N * 16 * 4, stream);

    node_h_kernel<<<(N + 255) / 256, 256, 0, stream>>>(x, W_gat, att_src, att_dst,
                                                       h, a_src, a_dst, N);
    weff_kernel<<<1, 32, 0, stream>>>(W_edge, att_edge, weff);
    alpha_max_kernel<<<nb, 256, 0, stream>>>(src, dst, edge_attr, a_src, a_dst,
                                             weff, alpha, nodemax, E);
    exp_sum_kernel<<<nb, 256, 0, stream>>>(dst, alpha, nodemax, denom, E);
    aggregate_kernel<<<nb, 256, 0, stream>>>(src, dst, alpha, denom, h, acc, E);
    relu_bias_kernel<<<(N * 16 + 255) / 256, 256, 0, stream>>>(acc, b_gat, xh, N * 16);

    int ntiles  = (E + 15) / 16;
    int nblocks = (ntiles + 7) / 8;
    edge_mlp_wmma_kernel<<<nblocks, 256, 0, stream>>>(xh, src, dst, edge_attr,
                                                      W1, b1, W2, b2, ef, E);

    count_kernel<<<nb, 256, 0, stream>>>(src, dst, lab, bcounts, E);
    scan_kernel<<<1, 32, 0, stream>>>(bcounts, boffs, nb);
    compact_kernel<<<nb, 256, 0, stream>>>(src, dst, lab, ef, boffs, out, E, K);
}
